// LFM2HybridArchitecture_6725918785578
// MI455X (gfx1250) — compile-verified, matrix operand order checked by reading, not run on hardware
//
#include <hip/hip_runtime.h>

// ---------------------------------------------------------------------------
// Model dims
// ---------------------------------------------------------------------------
#define BD 2
#define SD 1024
#define HD 2048
#define CDD 512
#define VD 50000
#define NB_CONV 10
#define NB_ATTN 6
#define NHEADS 32
#define HEADD 64
#define GATE_S 0.8f
#define DT 0.1f

// ---------------------------------------------------------------------------
// WMMA types (CDNA5 gfx1250, wave32)
// ---------------------------------------------------------------------------
typedef __bf16 bf16_t;
typedef bf16_t v16bf __attribute__((ext_vector_type(16)));
typedef float  v8f   __attribute__((ext_vector_type(8)));

__device__ __forceinline__ bf16_t f2bf(float f) {
    union { float f; unsigned u; } in; in.f = f;
    unsigned r = in.u + 0x7FFFu + ((in.u >> 16) & 1u);
    union { unsigned short s; bf16_t b; } out;
    out.s = (unsigned short)(r >> 16);
    return out.b;
}

// Two fp32 -> one u32 {bf16(hi),bf16(lo)}: bias-round then single v_perm_b32.
__device__ __forceinline__ unsigned pack_bf2(float hi, float lo) {
    union { float f; unsigned u; } a, b; a.f = hi; b.f = lo;
    return __builtin_amdgcn_perm(a.u + 0x8000u, b.u + 0x8000u, 0x07060302u);
}

__device__ __forceinline__ float geluf(float x) {
    float c = 0.7978845608028654f * (x + 0.044715f * x * x * x);
    return 0.5f * x * (1.0f + tanhf(c));
}
__device__ __forceinline__ float sigmoidf(float x) { return 1.0f / (1.0f + expf(-x)); }

// LDS tile: 128 rows x 32 bf16, row stride 48 ushorts (96B, 16B-aligned rows)
#define LDK 48

// 16x32 bf16 WMMA operand fragment (ISA 7.12.2 layout):
//   half 0 (lanes 0-15):  K = {0..7, 16..23}
//   half 1 (lanes 16-31): K = {8..15, 24..31}
// -> two contiguous 8-elem (16B) runs per lane: ds_load_b128 x2.
__device__ __forceinline__ v16bf load_frag(const unsigned short* rowu, int half) {
    const bf16_t* row = (const bf16_t*)rowu;
    v16bf f;
#pragma unroll
    for (int i = 0; i < 16; ++i) {
        int k = (i & 7) + ((i >> 3) << 4) + half * 8;
        f[i] = row[k];
    }
    return f;
}

// ---------------------------------------------------------------------------
// Generic batched strided GEMM:  C = act(scale*(A * W^T + bias)) [+resid] [+=C]
//   A[m,k] at A + zoff + m*ars + k*acs ; W[n,k] at Bw + zoff + n*brs + k*bcs
//   C[m,n] at C + zoff + m*ldc + n
// Block tile 128x128, K-step 32, 8 waves (4 in M x 2 in N), wave tile 32x64.
// Fast path: register double-buffered float4 staging so next tile's global
// loads overlap this tile's fragment loads + 8 WMMAs.
// ---------------------------------------------------------------------------
struct GemmP {
    const float* A; const float* Bw; const float* bias; const float* resid; float* C;
    int M, N, K;
    int ars, acs, brs, bcs, ldc;
    int inner;
    int Ain, Aout, Bin, Bout, Cin, Cout;
    float scale;
    int act;        // 0 none, 1 gelu, 2 sigmoid, 3 tanh
    int accum;      // 1: C += result
};

__global__ __launch_bounds__(256) void gemm_wmma_kernel(GemmP p) {
    __shared__ unsigned short As[128][LDK];
    __shared__ unsigned short Bs[128][LDK];

    const int tid  = threadIdx.x;
    const int lane = tid & 31;
    const int wave = tid >> 5;
    const int wm   = wave & 3;      // 4 M-subtiles of 32
    const int wn   = wave >> 2;     // 2 N-subtiles of 64
    const int half = lane >> 4;
    const int lm   = lane & 15;

    const int bz = blockIdx.z;
    const int offA = (bz % p.inner) * p.Ain + (bz / p.inner) * p.Aout;
    const int offB = (bz % p.inner) * p.Bin + (bz / p.inner) * p.Bout;
    const int offC = (bz % p.inner) * p.Cin + (bz / p.inner) * p.Cout;

    const int m0 = blockIdx.y * 128;
    const int n0 = blockIdx.x * 128;

    v8f acc[2][4];
#pragma unroll
    for (int a = 0; a < 2; ++a)
#pragma unroll
        for (int t = 0; t < 4; ++t)
            acc[a][t] = (v8f){0.f, 0.f, 0.f, 0.f, 0.f, 0.f, 0.f, 0.f};

    const bool full = (m0 + 128 <= p.M) && (n0 + 128 <= p.N) &&
                      ((p.K & 31) == 0) && (p.acs == 1) && (p.bcs == 1);

    if (full) {
        // Each thread owns 4 rows (stride 32) x one float4 column slot per
        // operand; pointers advance 32 floats per K-step.
        const int r0 = tid >> 3;          // 0..31
        const int c4 = (tid & 7) * 4;     // 0,4,..,28
        const float* pA = p.A  + offA + (m0 + r0) * p.ars + c4;
        const float* pB = p.Bw + offB + (n0 + r0) * p.brs + c4;
        const int aStep = 32 * p.ars;
        const int bStep = 32 * p.brs;
        unsigned* Au = (unsigned*)&As[0][0];
        unsigned* Bu = (unsigned*)&Bs[0][0];
        const int lrow = LDK / 2;
        const int lbase = r0 * lrow + (c4 >> 1);
        const int nsteps = p.K >> 5;

        // Prologue: fetch K-tile 0 into registers.
        float4 ra[4], rb[4];
#pragma unroll
        for (int it = 0; it < 4; ++it) {
            ra[it] = *(const float4*)(pA + it * aStep);
            rb[it] = *(const float4*)(pB + it * bStep);
        }

        for (int s = 0; s < nsteps; ++s) {
            // Pack current K-tile regs -> LDS (bf16).
#pragma unroll
            for (int it = 0; it < 4; ++it) {
                int li = lbase + it * 32 * lrow;
                Au[li]     = pack_bf2(ra[it].y, ra[it].x);
                Au[li + 1] = pack_bf2(ra[it].w, ra[it].z);
                Bu[li]     = pack_bf2(rb[it].y, rb[it].x);
                Bu[li + 1] = pack_bf2(rb[it].w, rb[it].z);
            }
            __syncthreads();

            // Issue next tile's global loads now; they complete while the
            // fragment loads + WMMAs below execute. Branchless last-iter
            // handling: re-read the same tile (in bounds, result unused).
            const int adv = (s + 1 < nsteps) ? 32 : 0;
            pA += adv; pB += adv;
#pragma unroll
            for (int it = 0; it < 4; ++it) {
                ra[it] = *(const float4*)(pA + it * aStep);
                rb[it] = *(const float4*)(pB + it * bStep);
            }
            __builtin_prefetch(pA + 32, 0, 1);   // warm K-tile s+2
            __builtin_prefetch(pB + 32, 0, 1);

            v16bf af0 = load_frag(&As[wm * 32 + lm][0], half);
            v16bf af1 = load_frag(&As[wm * 32 + 16 + lm][0], half);
            v16bf bf0 = load_frag(&Bs[wn * 64 + lm][0], half);
            v16bf bf1 = load_frag(&Bs[wn * 64 + 16 + lm][0], half);
            v16bf bf2 = load_frag(&Bs[wn * 64 + 32 + lm][0], half);
            v16bf bf3 = load_frag(&Bs[wn * 64 + 48 + lm][0], half);

            acc[0][0] = __builtin_amdgcn_wmma_f32_16x16x32_bf16(false, af0, false, bf0, (short)0, acc[0][0], false, false);
            acc[0][1] = __builtin_amdgcn_wmma_f32_16x16x32_bf16(false, af0, false, bf1, (short)0, acc[0][1], false, false);
            acc[0][2] = __builtin_amdgcn_wmma_f32_16x16x32_bf16(false, af0, false, bf2, (short)0, acc[0][2], false, false);
            acc[0][3] = __builtin_amdgcn_wmma_f32_16x16x32_bf16(false, af0, false, bf3, (short)0, acc[0][3], false, false);
            acc[1][0] = __builtin_amdgcn_wmma_f32_16x16x32_bf16(false, af1, false, bf0, (short)0, acc[1][0], false, false);
            acc[1][1] = __builtin_amdgcn_wmma_f32_16x16x32_bf16(false, af1, false, bf1, (short)0, acc[1][1], false, false);
            acc[1][2] = __builtin_amdgcn_wmma_f32_16x16x32_bf16(false, af1, false, bf2, (short)0, acc[1][2], false, false);
            acc[1][3] = __builtin_amdgcn_wmma_f32_16x16x32_bf16(false, af1, false, bf3, (short)0, acc[1][3], false, false);
            __syncthreads();
        }
    } else {
        // Guarded path: boundary blocks / strided-K operands only.
        for (int k0 = 0; k0 < p.K; k0 += 32) {
#pragma unroll
            for (int it = 0; it < 16; ++it) {
                int idx = tid + it * 256;          // 0..4095
                int r = idx >> 5, c = idx & 31;
                int gk = k0 + c;
                float va = 0.f, vb = 0.f;
                int gm = m0 + r, gn = n0 + r;
                if (gm < p.M && gk < p.K) va = p.A[offA + gm * p.ars + gk * p.acs];
                if (gn < p.N && gk < p.K) vb = p.Bw[offB + gn * p.brs + gk * p.bcs];
                ((bf16_t*)&As[r][0])[c] = f2bf(va);
                ((bf16_t*)&Bs[r][0])[c] = f2bf(vb);
            }
            __syncthreads();

            v16bf af0 = load_frag(&As[wm * 32 + lm][0], half);
            v16bf af1 = load_frag(&As[wm * 32 + 16 + lm][0], half);
#pragma unroll
            for (int t = 0; t < 4; ++t) {
                v16bf bf = load_frag(&Bs[wn * 64 + t * 16 + lm][0], half);
                acc[0][t] = __builtin_amdgcn_wmma_f32_16x16x32_bf16(false, af0, false, bf, (short)0, acc[0][t], false, false);
                acc[1][t] = __builtin_amdgcn_wmma_f32_16x16x32_bf16(false, af1, false, bf, (short)0, acc[1][t], false, false);
            }
            __syncthreads();
        }
    }

    // Epilogue. C/D layout: lane(lm)=N, vgpr j -> M = j + 8*half.
#pragma unroll
    for (int a = 0; a < 2; ++a) {
#pragma unroll
        for (int t = 0; t < 4; ++t) {
            int col = n0 + wn * 64 + t * 16 + lm;
            if (col >= p.N) continue;
            float bval = p.bias ? p.bias[col] : 0.f;
#pragma unroll
            for (int j = 0; j < 8; ++j) {
                int row = m0 + wm * 32 + a * 16 + half * 8 + j;
                if (row >= p.M) continue;
                float v = p.scale * (acc[a][t][j] + bval);
                if (p.act == 1)      v = geluf(v);
                else if (p.act == 2) v = sigmoidf(v);
                else if (p.act == 3) v = tanhf(v);
                int ci = offC + row * p.ldc + col;
                if (p.resid) v += p.resid[ci];
                if (p.accum) v += p.C[ci];
                p.C[ci] = v;
            }
        }
    }
}

// ---------------------------------------------------------------------------
// Embedding gather + broadcast context projection
// ---------------------------------------------------------------------------
__global__ void embed_add_kernel(const int* __restrict__ ids,
                                 const float* __restrict__ emb,
                                 const float* __restrict__ ctxp,
                                 float* __restrict__ x) {
    long long e = (long long)blockIdx.x * blockDim.x + threadIdx.x;
    if (e >= (long long)BD * SD * HD) return;
    int h = (int)(e % HD);
    long long bs = e / HD;
    int b = (int)(bs / SD);
    int tok = ids[bs];
    x[e] = emb[(long long)tok * HD + h] + ctxp[b * HD + h];
}

// ---------------------------------------------------------------------------
// Grouped conv1d, kernel 3, SAME padding, 8 channels/group
// ---------------------------------------------------------------------------
__global__ void conv_kernel(const float* __restrict__ x,
                            const float* __restrict__ w,
                            const float* __restrict__ bias,
                            float* __restrict__ xc) {
    long long e = (long long)blockIdx.x * blockDim.x + threadIdx.x;
    if (e >= (long long)BD * SD * HD) return;
    int oc = (int)(e % HD);
    long long bs = e / HD;
    int s = (int)(bs % SD);
    int b = (int)(bs / SD);
    int g = oc >> 3;
    const float* wr = w + (long long)oc * 24;   // [oc][8][3]
    float acc = bias[oc];
    const float* xb = x + ((long long)b * SD) * HD;
#pragma unroll
    for (int t = 0; t < 3; ++t) {
        int ss = s + t - 1;
        if (ss < 0 || ss >= SD) continue;
        const float* xr = xb + (long long)ss * HD + g * 8;
#pragma unroll
        for (int ic = 0; ic < 8; ++ic)
            acc += wr[ic * 3 + t] * xr[ic];
    }
    xc[e] = acc;
}

// ---------------------------------------------------------------------------
// Fused combine + LayerNorm. mode 0: t = a+b ; mode 1: t = a*b + c*d*GATE_S
// ---------------------------------------------------------------------------
__global__ __launch_bounds__(256) void fuse_ln_kernel(const float* __restrict__ a,
                                                      const float* __restrict__ b,
                                                      const float* __restrict__ c,
                                                      const float* __restrict__ d,
                                                      const float* __restrict__ g,
                                                      const float* __restrict__ beta,
                                                      float* __restrict__ out, int mode) {
    __shared__ float red[256];
    const int tid = threadIdx.x;
    const long long base = (long long)blockIdx.x * HD;
    float tv[8];
    float s = 0.f;
#pragma unroll
    for (int i = 0; i < 8; ++i) {
        int idx = tid + i * 256;
        float t;
        if (mode == 0) t = a[base + idx] + b[base + idx];
        else           t = a[base + idx] * b[base + idx] +
                           c[base + idx] * d[base + idx] * GATE_S;
        tv[i] = t; s += t;
    }
    red[tid] = s; __syncthreads();
    for (int st = 128; st > 0; st >>= 1) {
        if (tid < st) red[tid] += red[tid + st];
        __syncthreads();
    }
    float mean = red[0] / (float)HD;
    __syncthreads();
    float s2 = 0.f;
#pragma unroll
    for (int i = 0; i < 8; ++i) { float dd = tv[i] - mean; s2 += dd * dd; }
    red[tid] = s2; __syncthreads();
    for (int st = 128; st > 0; st >>= 1) {
        if (tid < st) red[tid] += red[tid + st];
        __syncthreads();
    }
    float inv = rsqrtf(red[0] / (float)HD + 1e-5f);
#pragma unroll
    for (int i = 0; i < 8; ++i) {
        int idx = tid + i * 256;
        out[base + idx] = (tv[i] - mean) * inv * g[idx] + beta[idx];
    }
}

// ---------------------------------------------------------------------------
// Row softmax over S=1024 columns, in place. One block per row.
// ---------------------------------------------------------------------------
__global__ __launch_bounds__(256) void softmax_kernel(float* __restrict__ sc) {
    __shared__ float red[256];
    const int tid = threadIdx.x;
    const long long base = (long long)blockIdx.x * SD;
    float tv[4];
    float mx = -3.4e38f;
#pragma unroll
    for (int i = 0; i < 4; ++i) {
        tv[i] = sc[base + tid + i * 256];
        mx = fmaxf(mx, tv[i]);
    }
    red[tid] = mx; __syncthreads();
    for (int st = 128; st > 0; st >>= 1) {
        if (tid < st) red[tid] = fmaxf(red[tid], red[tid + st]);
        __syncthreads();
    }
    mx = red[0];
    __syncthreads();
    float s = 0.f;
#pragma unroll
    for (int i = 0; i < 4; ++i) { tv[i] = expf(tv[i] - mx); s += tv[i]; }
    red[tid] = s; __syncthreads();
    for (int st = 128; st > 0; st >>= 1) {
        if (tid < st) red[tid] += red[tid + st];
        __syncthreads();
    }
    float inv = 1.0f / red[0];
#pragma unroll
    for (int i = 0; i < 4; ++i) sc[base + tid + i * 256] = tv[i] * inv;
}

// ---------------------------------------------------------------------------
// LTC sequential Euler scan. One block per batch, 512 threads (one / channel),
// hidden state in LDS; weight rows streamed as float4 (L2-resident, 3 MB).
// ---------------------------------------------------------------------------
__global__ __launch_bounds__(512) void ltc_scan_kernel(const float* __restrict__ pre_in,
                                                       const float* __restrict__ rec_w,
                                                       const float* __restrict__ rec_b,
                                                       const float* __restrict__ cg_w,
                                                       const float* __restrict__ cg_b,
                                                       const float* __restrict__ eg_w,
                                                       const float* __restrict__ eg_b,
                                                       const float* __restrict__ tau,
                                                       float* __restrict__ hseq) {
    __shared__ float h[CDD];
    const int j = threadIdx.x;
    const int b = blockIdx.x;
    h[j] = 0.f;
    __syncthreads();
    const float4* rw = (const float4*)(rec_w + (long long)j * CDD);
    const float4* cw = (const float4*)(cg_w + (long long)j * CDD);
    const float4* ew = (const float4*)(eg_w + (long long)j * CDD);
    const float rb = rec_b[j], cb = cg_b[j], ebi = eg_b[j];
    const float tj = tau[j];
    for (int t = 0; t < SD; ++t) {
        float aP = rb, aC = cb, aE = ebi;
        for (int k4 = 0; k4 < CDD / 4; ++k4) {
            float4 hh = *(const float4*)&h[k4 * 4];
            float4 a = rw[k4];
            aP += a.x * hh.x + a.y * hh.y + a.z * hh.z + a.w * hh.w;
            float4 c = cw[k4];
            aC += c.x * hh.x + c.y * hh.y + c.z * hh.z + c.w * hh.w;
            float4 e = ew[k4];
            aE += e.x * hh.x + e.y * hh.y + e.z * hh.z + e.w * hh.w;
        }
        long long ri = ((long long)b * SD + t) * CDD + j;
        float pre = fmaxf(pre_in[ri] + aP, 0.f);
        float cg = sigmoidf(aC);
        float eg = tanhf(aE);
        float hj = h[j];
        float dh = (-hj + pre) / tj;
        dh = dh * cg + 0.1f * eg;
        float hn = hj + DT * dh;
        __syncthreads();
        h[j] = hn;
        hseq[ri] = hn;
        __syncthreads();
    }
}

// ---------------------------------------------------------------------------
// Host side
// ---------------------------------------------------------------------------
static void launch_gemm(hipStream_t st, const GemmP& p, int batches) {
    dim3 grid((p.N + 127) / 128, (p.M + 127) / 128, batches);
    hipLaunchKernelGGL(gemm_wmma_kernel, grid, dim3(256), 0, st, p);
}

static GemmP mkgemm(const float* A, const float* W, const float* bias,
                    const float* resid, float* C, int M, int N, int K,
                    int ars, int brs, int ldc, float scale, int act, int accum) {
    GemmP p{};
    p.A = A; p.Bw = W; p.bias = bias; p.resid = resid; p.C = C;
    p.M = M; p.N = N; p.K = K;
    p.ars = ars; p.acs = 1; p.brs = brs; p.bcs = 1; p.ldc = ldc;
    p.inner = 1;
    p.Ain = p.Aout = p.Bin = p.Bout = p.Cin = p.Cout = 0;
    p.scale = scale; p.act = act; p.accum = accum;
    return p;
}

extern "C" void kernel_launch(void* const* d_in, const int* in_sizes, int n_in,
                              void* d_out, int out_size, void* d_ws, size_t ws_size,
                              hipStream_t stream) {
    const int*   input_ids = (const int*)  d_in[0];
    const float* ctx       = (const float*)d_in[1];
    const float* emb       = (const float*)d_in[2];
    const float* ce_w      = (const float*)d_in[3];
    const float* ce_b      = (const float*)d_in[4];
    const float* conv_w    = (const float*)d_in[5];
    const float* conv_b    = (const float*)d_in[6];
    const float* enh_w     = (const float*)d_in[7];
    const float* enh_b     = (const float*)d_in[8];
    const float* ln1_g     = (const float*)d_in[9];
    const float* ln1_b     = (const float*)d_in[10];
    const float* gi_w      = (const float*)d_in[11];
    const float* gi_b      = (const float*)d_in[12];
    const float* gf_w      = (const float*)d_in[13];
    const float* gf_b      = (const float*)d_in[14];
    const float* gc_w      = (const float*)d_in[15];
    const float* gc_b      = (const float*)d_in[16];
    const float* ln2_g     = (const float*)d_in[17];
    const float* ln2_b     = (const float*)d_in[18];
    const float* ain_w     = (const float*)d_in[19];
    const float* ain_b     = (const float*)d_in[20];
    const float* aout_w    = (const float*)d_in[21];
    const float* aout_b    = (const float*)d_in[22];
    const float* tau       = (const float*)d_in[23];
    const float* ltc_in_w  = (const float*)d_in[24];
    const float* ltc_in_b  = (const float*)d_in[25];
    const float* ltc_rec_w = (const float*)d_in[26];
    const float* ltc_rec_b = (const float*)d_in[27];
    const float* ltc_out_w = (const float*)d_in[28];
    const float* ltc_out_b = (const float*)d_in[29];
    const float* ltc_cg_w  = (const float*)d_in[30];
    const float* ltc_cg_b  = (const float*)d_in[31];
    const float* ltc_eg_w  = (const float*)d_in[32];
    const float* ltc_eg_b  = (const float*)d_in[33];
    const float* cp_w      = (const float*)d_in[34];
    const float* cp_b      = (const float*)d_in[35];
    const float* op_w      = (const float*)d_in[36];
    const float* op_b      = (const float*)d_in[37];
    const float* emp_w     = (const float*)d_in[38];
    const float* emp_b     = (const float*)d_in[39];
    const float* cre_w     = (const float*)d_in[40];
    const float* cre_b     = (const float*)d_in[41];

    float* logits = (float*)d_out;                                // [B,S,V]
    float* cf_out = (float*)d_out + (size_t)BD * SD * VD;         // [B,S,CD]

    float* w = (float*)d_ws;
    size_t off = 0;
    const size_t BS = (size_t)BD * SD;
    auto carve = [&](size_t n) { float* p = w + off; off += n; return p; };
    float* x    = carve(BS * HD);
    float* xc   = carve(BS * HD);
    float* bufA = carve(BS * HD);
    float* bufB = carve(BS * HD);
    float* bufC = carve(BS * HD);
    float* qkv  = carve(BS * 3 * HD);
    float* scor = carve((size_t)BD * NHEADS * SD * SD);
    float* ctxp = carve((size_t)BD * HD);
    float* prei = carve(BS * CDD);
    float* hseq = carve(BS * CDD);
    float* comb = carve(BS * HD);
    float* cf   = carve(BS * CDD);
    (void)ws_size; (void)in_sizes; (void)n_in; (void)out_size;

    const int M = (int)BS;
    const long long totE = (long long)BS * HD;
    const int eb = (int)((totE + 255) / 256);

    // 1. context projection
    launch_gemm(stream, mkgemm(ctx, ce_w, ce_b, nullptr, ctxp,
                               BD, HD, CDD, CDD, CDD, HD, 1.f, 0, 0), 1);
    // 2. embedding + broadcast add
    hipLaunchKernelGGL(embed_add_kernel, dim3(eb), dim3(256), 0, stream,
                       input_ids, emb, ctxp, x);

    // 3. conv blocks
    for (int i = 0; i < NB_CONV; ++i) {
        hipLaunchKernelGGL(conv_kernel, dim3(eb), dim3(256), 0, stream,
                           x, conv_w + (size_t)i * HD * 24, conv_b + (size_t)i * HD, xc);
        launch_gemm(stream, mkgemm(xc, enh_w + (size_t)i * HD * HD, enh_b + (size_t)i * HD,
                                   nullptr, bufA, M, HD, HD, HD, HD, HD, 1.f, 1, 0), 1);
        hipLaunchKernelGGL(fuse_ln_kernel, dim3(M), dim3(256), 0, stream,
                           x, bufA, nullptr, nullptr,
                           ln1_g + (size_t)i * HD, ln1_b + (size_t)i * HD, x, 0);
        launch_gemm(stream, mkgemm(x, gi_w + (size_t)i * HD * HD, gi_b + (size_t)i * HD,
                                   nullptr, bufA, M, HD, HD, HD, HD, HD, 1.f, 2, 0), 1);
        launch_gemm(stream, mkgemm(x, gf_w + (size_t)i * HD * HD, gf_b + (size_t)i * HD,
                                   nullptr, bufB, M, HD, HD, HD, HD, HD, 1.f, 2, 0), 1);
        launch_gemm(stream, mkgemm(x, gc_w + (size_t)i * HD * HD, gc_b + (size_t)i * HD,
                                   nullptr, bufC, M, HD, HD, HD, HD, HD, 1.f, 3, 0), 1);
        hipLaunchKernelGGL(fuse_ln_kernel, dim3(M), dim3(256), 0, stream,
                           x, bufB, bufA, bufC,
                           ln2_g + (size_t)i * HD, ln2_b + (size_t)i * HD, x, 1);
    }

    // 4. attention blocks
    for (int i = 0; i < NB_ATTN; ++i) {
        launch_gemm(stream, mkgemm(x, ain_w + (size_t)i * 3 * HD * HD,
                                   ain_b + (size_t)i * 3 * HD,
                                   nullptr, qkv, M, 3 * HD, HD, HD, HD, 3 * HD, 1.f, 0, 0), 1);
        // scores[b,h] = Q K^T / 8
        GemmP ps{};
        ps.A = qkv; ps.Bw = qkv + HD; ps.bias = nullptr; ps.resid = nullptr; ps.C = scor;
        ps.M = SD; ps.N = SD; ps.K = HEADD;
        ps.ars = 3 * HD; ps.acs = 1; ps.brs = 3 * HD; ps.bcs = 1; ps.ldc = SD;
        ps.inner = NHEADS;
        ps.Ain = HEADD;      ps.Aout = SD * 3 * HD;
        ps.Bin = HEADD;      ps.Bout = SD * 3 * HD;
        ps.Cin = SD * SD;    ps.Cout = NHEADS * SD * SD;
        ps.scale = 0.125f; ps.act = 0; ps.accum = 0;
        launch_gemm(stream, ps, BD * NHEADS);
        hipLaunchKernelGGL(softmax_kernel, dim3(BD * NHEADS * SD), dim3(256), 0, stream, scor);
        // o[b,h] = att @ V  (V transposed via stride swap)
        GemmP pv{};
        pv.A = scor; pv.Bw = qkv + 2 * HD; pv.bias = nullptr; pv.resid = nullptr; pv.C = bufA;
        pv.M = SD; pv.N = HEADD; pv.K = SD;
        pv.ars = SD; pv.acs = 1; pv.brs = 1; pv.bcs = 3 * HD; pv.ldc = HD;
        pv.inner = NHEADS;
        pv.Ain = SD * SD;    pv.Aout = NHEADS * SD * SD;
        pv.Bin = HEADD;      pv.Bout = SD * 3 * HD;
        pv.Cin = HEADD;      pv.Cout = SD * HD;
        pv.scale = 1.f; pv.act = 0; pv.accum = 0;
        launch_gemm(stream, pv, BD * NHEADS);
        launch_gemm(stream, mkgemm(bufA, aout_w + (size_t)i * HD * HD,
                                   aout_b + (size_t)i * HD,
                                   x, x, M, HD, HD, HD, HD, HD, 1.f, 0, 0), 1);
    }

    // 5. LTC input projection
    launch_gemm(stream, mkgemm(x, ltc_in_w, ltc_in_b, nullptr, prei,
                               M, CDD, HD, HD, HD, CDD, 1.f, 0, 0), 1);
    // 6. sequential scan
    hipLaunchKernelGGL(ltc_scan_kernel, dim3(BD), dim3(CDD), 0, stream,
                       prei, ltc_rec_w, ltc_rec_b, ltc_cg_w, ltc_cg_b,
                       ltc_eg_w, ltc_eg_b, tau, hseq);
    // 7. combined = x + hseq @ ltc_out_w^T + b
    launch_gemm(stream, mkgemm(hseq, ltc_out_w, ltc_out_b, x, comb,
                               M, HD, CDD, CDD, CDD, HD, 1.f, 0, 0), 1);
    // 8. cf = combined @ cp_w^T + cp_b
    launch_gemm(stream, mkgemm(comb, cp_w, cp_b, nullptr, cf,
                               M, CDD, HD, HD, HD, CDD, 1.f, 0, 0), 1);
    // 9. cf_out = 0.6*emp + 0.4*cre
    launch_gemm(stream, mkgemm(cf, emp_w, emp_b, nullptr, cf_out,
                               M, CDD, CDD, CDD, CDD, CDD, 0.6f, 0, 0), 1);
    launch_gemm(stream, mkgemm(cf, cre_w, cre_b, nullptr, cf_out,
                               M, CDD, CDD, CDD, CDD, CDD, 0.4f, 0, 1), 1);
    // 10. logits
    launch_gemm(stream, mkgemm(comb, op_w, op_b, nullptr, logits,
                               M, VD, HD, HD, HD, VD, 1.f, 0, 0), 1);
}
